// SAGPool_79972291052238
// MI455X (gfx1250) — compile-verified, matrix operand order checked by reading, not run on hardware
//
#include <hip/hip_runtime.h>
#include <hip/hip_bf16.h>
#include <math.h>

#define BB   256
#define NPG  2048
#define DD   128
#define DEGC 16
#define NN   (BB * NPG)          // 524288 nodes
#define EE   (NN * DEGC)         // 8388608 edges
#define KK   (NPG / 2)           // 1024 kept per graph

typedef float v2f  __attribute__((ext_vector_type(2)));
typedef float v8f  __attribute__((ext_vector_type(8)));
typedef _Float16 v16h __attribute__((ext_vector_type(16)));

// ---------------------------------------------------------------------------
// K0: init deg=1.0 (self-loop), keep=0, new_id=0
// ---------------------------------------------------------------------------
__global__ void init_kernel(float* __restrict__ deg, int* __restrict__ keep,
                            int* __restrict__ nid) {
    int i = blockIdx.x * blockDim.x + threadIdx.x;
    if (i >= NN) return;
    deg[i]  = 1.0f;
    keep[i] = 0;
    nid[i]  = 0;
}

// ---------------------------------------------------------------------------
// K1: deg[dst] += w   (segment_sum of edge weights)
// ---------------------------------------------------------------------------
__global__ void deg_kernel(const int* __restrict__ edst,
                           const float* __restrict__ ew,
                           float* __restrict__ deg) {
    int e = blockIdx.x * blockDim.x + threadIdx.x;
    if (e >= EE) return;
    atomicAdd(&deg[edst[e]], ew[e]);
}

// ---------------------------------------------------------------------------
// K2: h = x @ W via WMMA. One wave per 16 rows.
// ---------------------------------------------------------------------------
__global__ void matvec_wmma(const float* __restrict__ x,
                            const float* __restrict__ W,
                            float* __restrict__ h) {
    int wave = (blockIdx.x * blockDim.x + threadIdx.x) >> 5;
    int lane = threadIdx.x & 31;
    int base = wave * 16;
    if (base >= NN) return;
    int m    = lane & 15;
    int half = lane >> 4;
    const float* row = x + (size_t)(base + m) * DD;

    v8f c = {};
#if __has_builtin(__builtin_amdgcn_wmma_f32_16x16x4_f32)
    // f32 A 16x4 layout: VGPR0 = K=(0|2) per half-wave, VGPR1 = K=(1|3).
    for (int kk = 0; kk < DD; kk += 4) {
        v2f a, b;
        a.x = row[kk + 2 * half + 0];
        a.y = row[kk + 2 * half + 1];
        // Broadcast W[k] across all 16 B-columns -> every C column = dot(x_row, W)
        b.x = W[kk + 2 * half + 0];
        b.y = W[kk + 2 * half + 1];
        c = __builtin_amdgcn_wmma_f32_16x16x4_f32(false, a, false, b,
                                                  (short)0, c, false, false);
    }
#else
    // Fallback: f16 inputs, f32 accumulate (probe-confirmed builtin).
    for (int kk = 0; kk < DD; kk += 32) {
        v16h a, b;
#pragma unroll
        for (int p = 0; p < 8; ++p) {
            int k = (p < 4) ? (half * 8 + 2 * p) : (16 + half * 8 + 2 * (p - 4));
            a[2 * p + 0] = (_Float16)row[kk + k + 0];
            a[2 * p + 1] = (_Float16)row[kk + k + 1];
            b[2 * p + 0] = (_Float16)W[kk + k + 0];
            b[2 * p + 1] = (_Float16)W[kk + k + 1];
        }
        c = __builtin_amdgcn_wmma_f32_16x16x32_f16(false, a, false, b,
                                                   (short)0, c, false, false);
    }
#endif
    // C layout: lane (half*16) column N=0 holds M = half*8 + v in c[v].
    if (m == 0) {
        float* out = h + base + 8 * half;
#pragma unroll
        for (int v = 0; v < 8; ++v) out[v] = c[v];
    }
}

// ---------------------------------------------------------------------------
// K3: score = h/deg (self-loop term); overwrite deg with rsqrt(deg)
// ---------------------------------------------------------------------------
__global__ void selfterm_kernel(const float* __restrict__ h,
                                float* __restrict__ deg_then_rsq,
                                float* __restrict__ sc) {
    int i = blockIdx.x * blockDim.x + threadIdx.x;
    if (i >= NN) return;
    float d = deg_then_rsq[i];
    float r = rsqrtf(d);
    sc[i] = h[i] * r * r;          // h / deg
    deg_then_rsq[i] = r;           // now holds inv_sqrt(deg)
}

// ---------------------------------------------------------------------------
// K4: sc[dst] += w * rsq[src] * rsq[dst] * h[src]
// ---------------------------------------------------------------------------
__global__ void edge_acc_kernel(const int* __restrict__ esrc,
                                const int* __restrict__ edst,
                                const float* __restrict__ ew,
                                const float* __restrict__ h,
                                const float* __restrict__ rsq,
                                float* __restrict__ sc) {
    int e = blockIdx.x * blockDim.x + threadIdx.x;
    if (e >= EE) return;
    int s = esrc[e], d = edst[e];
    atomicAdd(&sc[d], ew[e] * rsq[s] * rsq[d] * h[s]);
}

// ---------------------------------------------------------------------------
// K5: exact per-graph top-K via LDS bitonic sort (matches lax.top_k order:
//     descending score, stable ascending index on ties).
// ---------------------------------------------------------------------------
__global__ void __launch_bounds__(1024)
topk_kernel(const float* __restrict__ sc, int* __restrict__ tki,
            int* __restrict__ keep, int* __restrict__ nid) {
    __shared__ unsigned long long keys[NPG];
    int g = blockIdx.x;
    int t = threadIdx.x;
    const float* s = sc + (size_t)g * NPG;

    for (int i = t; i < NPG; i += 1024) {
        unsigned int fb  = __float_as_uint(s[i]);
        unsigned int ord = (fb & 0x80000000u) ? ~fb : (fb ^ 0x80000000u); // asc in value
        // key ascending => score descending, tie -> smaller index first
        keys[i] = ((unsigned long long)(~ord) << 11) | (unsigned long long)i;
    }
    __syncthreads();

    for (int k = 2; k <= NPG; k <<= 1) {
        for (int j = k >> 1; j > 0; j >>= 1) {
            for (int i = t; i < NPG; i += 1024) {
                int ixj = i ^ j;
                if (ixj > i) {
                    bool up = ((i & k) == 0);
                    unsigned long long a = keys[i], b = keys[ixj];
                    if ((a > b) == up) { keys[i] = b; keys[ixj] = a; }
                }
            }
            __syncthreads();
        }
    }

    if (t < KK) {
        int local  = (int)(keys[t] & 2047ull);
        int global = g * NPG + local;
        int pos    = g * KK + t;
        tki[pos]   = global;
        keep[global] = 1;
        nid[global]  = pos;
    }
}

// ---------------------------------------------------------------------------
// K6: pooled_x[i] = x[tki[i]] * tanh(score[tki[i]]) ; pooled graph index.
//     One wave per row, float4 per lane (32*16B = 128 floats).
// ---------------------------------------------------------------------------
__global__ void gather_kernel(const float* __restrict__ x,
                              const float* __restrict__ sc,
                              const int* __restrict__ tki,
                              const int* __restrict__ ngi,
                              float* __restrict__ out_px,
                              int* __restrict__ out_ngi) {
    int wave = (blockIdx.x * blockDim.x + threadIdx.x) >> 5;
    int lane = threadIdx.x & 31;
    if (wave >= BB * KK) return;
    int node = tki[wave];
    float gate = tanhf(sc[node]);
    const float4* src = (const float4*)(x + (size_t)node * DD);
    float4 v = src[lane];
    v.x *= gate; v.y *= gate; v.z *= gate; v.w *= gate;
    ((float4*)(out_px + (size_t)wave * DD))[lane] = v;
    if (lane == 0) out_ngi[wave] = ngi[node];
}

// ---------------------------------------------------------------------------
// K7: edge remap + mask
// ---------------------------------------------------------------------------
__global__ void edge_out_kernel(const int* __restrict__ esrc,
                                const int* __restrict__ edst,
                                const float* __restrict__ ew,
                                const int* __restrict__ keep,
                                const int* __restrict__ nid,
                                int* __restrict__ out_src,
                                int* __restrict__ out_dst,
                                float* __restrict__ out_w) {
    int e = blockIdx.x * blockDim.x + threadIdx.x;
    if (e >= EE) return;
    int s = esrc[e], d = edst[e];
    bool kp = (keep[s] != 0) && (keep[d] != 0);
    out_src[e] = kp ? nid[s] : -1;
    out_dst[e] = kp ? nid[d] : -1;
    out_w[e]   = kp ? ew[e] : 0.0f;
}

// ---------------------------------------------------------------------------
extern "C" void kernel_launch(void* const* d_in, const int* in_sizes, int n_in,
                              void* d_out, int out_size, void* d_ws, size_t ws_size,
                              hipStream_t stream) {
    const float* x   = (const float*)d_in[0];
    const int*   ei  = (const int*)d_in[1];
    const float* ew  = (const float*)d_in[2];
    const int*   ngi = (const int*)d_in[3];
    const float* W   = (const float*)d_in[4];
    const int* esrc = ei;
    const int* edst = ei + EE;

    // Workspace partitions (~11.5 MB)
    float* h    = (float*)d_ws;          // [NN]
    float* rsq  = h + NN;                // [NN] deg -> rsqrt(deg)
    float* sc   = rsq + NN;              // [NN]
    int*   keep = (int*)(sc + NN);       // [NN]
    int*   nid  = keep + NN;             // [NN]
    int*   tki  = nid + NN;              // [BB*KK]

    // Output partitions (flat concat, element-aligned 4B each)
    float* out_px  = (float*)d_out;                                  // [BB*KK*DD]
    int*   out_ei  = (int*)(out_px + (size_t)BB * KK * DD);          // [2*EE]
    float* out_ew  = out_px + (size_t)BB * KK * DD + 2 * (size_t)EE; // [EE]
    int*   out_ngi = (int*)(out_px + (size_t)BB * KK * DD + 3 * (size_t)EE); // [BB*KK]

    init_kernel   <<<NN / 256, 256, 0, stream>>>(rsq, keep, nid);
    deg_kernel    <<<EE / 256, 256, 0, stream>>>(edst, ew, rsq);
    matvec_wmma   <<<NN / 16 / 8, 256, 0, stream>>>(x, W, h);   // 8 waves/block
    selfterm_kernel<<<NN / 256, 256, 0, stream>>>(h, rsq, sc);
    edge_acc_kernel<<<EE / 256, 256, 0, stream>>>(esrc, edst, ew, h, rsq, sc);
    topk_kernel   <<<BB, 1024, 0, stream>>>(sc, tki, keep, nid);
    gather_kernel <<<BB * KK / 8, 256, 0, stream>>>(x, sc, tki, ngi, out_px, out_ngi);
    edge_out_kernel<<<EE / 256, 256, 0, stream>>>(esrc, edst, ew, keep, nid,
                                                  out_ei, out_ei + EE, out_ew);
}
